// Attention_79130477462096
// MI455X (gfx1250) — matrix operand order repaired and checked by reading, not compiled, not run on hardware
//
#include <hip/hip_runtime.h>

// ---------------------------------------------------------------------------
// Causal MHA forward for gfx1250 (MI455X), bf16 WMMA pipeline, f32 accumulate.
// - All fragments are aligned ds_load_b128 (B operands staged transposed).
// - Tile staging uses GLOBAL_LOAD_ASYNC_TO_LDS_B128 via inline asm (ASYNCcnt).
// - GEMM kernels: 128x64 block tiles, 4 waves x (32 rows x 64 cols).
// ---------------------------------------------------------------------------

#define DEVI __device__ __forceinline__

typedef __bf16 bf16;
typedef __attribute__((ext_vector_type(16))) __bf16 v16bf;
typedef __attribute__((ext_vector_type(4)))  __bf16 v4bf;
typedef __attribute__((ext_vector_type(8)))  float  v8f;

constexpr int BATCH = 2, SEQ = 2048, NH = 16, DH = 64, DM = 1024;
constexpr float MASK_NEG = -1.0e30f;

// workspace layout (bytes); total 40 MB. z reuses the x_bf16 region.
constexpr size_t OFF_X  = 0;                            // x bf16 [B*S][DM]   (8 MB)
constexpr size_t OFF_WQ = 8u  * 1024 * 1024;            // Wt [NH][DH][DM]    (2 MB)
constexpr size_t OFF_WK = 10u * 1024 * 1024;
constexpr size_t OFF_WV = 12u * 1024 * 1024;
constexpr size_t OFF_WO = 14u * 1024 * 1024;            // WoT [DM n][DM k]   (2 MB)
constexpr size_t OFF_Q  = 16u * 1024 * 1024;            // [(b*NH+h)*S+s][DH] (8 MB)
constexpr size_t OFF_K  = 24u * 1024 * 1024;
constexpr size_t OFF_V  = 32u * 1024 * 1024;            // [(b*NH+h)*DH+d][S] (transposed)
constexpr size_t OFF_Z  = OFF_X;                        // z bf16 [B*S][DM]   (reuse)

// ---------------------------------------------------------------------------
// Async global->LDS staging via inline asm (ISA §15.18: opcode 98, VGLOBAL).
// Each lane copies 16 bytes global->LDS (offset added to both addresses), ASYNCcnt.
// Generic pointers to LDS keep the LDS byte address in the low 32 bits.
// ---------------------------------------------------------------------------
template <int OFF>
DEVI void async_b128(const bf16* g, bf16* s) {
  const unsigned lds_addr = (unsigned)(unsigned long long)s;
  asm volatile("global_load_async_to_lds_b128 %0, %1, off offset:%2"
               :: "v"(lds_addr), "v"(g), "i"(OFF)
               : "memory");
}

DEVI void wait_async0() {
#if __has_builtin(__builtin_amdgcn_s_wait_asynccnt)
  __builtin_amdgcn_s_wait_asynccnt(0);
#else
  asm volatile("s_wait_asynccnt 0x0" ::: "memory");
#endif
}
#define WAIT_ASYNC() wait_async0()

// 128 threads copy a 64x64 bf16 tile (global ld = ldg) into LDS (ld = 72).
DEVI void copy64x64(const bf16* __restrict__ g, int ldg, bf16* __restrict__ s) {
  const int row = threadIdx.x >> 1;
  const int col = (threadIdx.x & 1) * 32;
  const bf16* gp = g + (size_t)row * ldg + col;
  bf16* sp = s + row * 72 + col;
  async_b128<0>(gp, sp);  async_b128<16>(gp, sp);
  async_b128<32>(gp, sp); async_b128<48>(gp, sp);
}

// 128 threads copy a 128x32 bf16 tile into LDS (ld = 32). One row per thread.
DEVI void copy128x32(const bf16* __restrict__ g, int ldg, bf16* __restrict__ s) {
  const int row = threadIdx.x;
  const bf16* gp = g + (size_t)row * ldg;
  bf16* sp = s + row * 32;
  async_b128<0>(gp, sp);  async_b128<16>(gp, sp);
  async_b128<32>(gp, sp); async_b128<48>(gp, sp);
}

// 128 threads copy a 64x32 bf16 tile into LDS (ld = lds_ld).
DEVI void copy64x32(const bf16* __restrict__ g, int ldg,
                    bf16* __restrict__ s, int lds_ld) {
  const int row = threadIdx.x >> 1;
  const int col = (threadIdx.x & 1) * 16;
  const bf16* gp = g + (size_t)row * ldg + col;
  bf16* sp = s + row * lds_ld + col;
  async_b128<0>(gp, sp);  async_b128<16>(gp, sp);
}

// ---------------------------------------------------------------------------
// WMMA fragment accessors (layouts per CDNA5 ISA §7.12.2).
// ---------------------------------------------------------------------------
DEVI v16bf frag2x16(const bf16* p0, const bf16* p1) {
  union { uint4 u[2]; v16bf v; } r;
  r.u[0] = *(const uint4*)p0;
  r.u[1] = *(const uint4*)p1;
  return r.v;
}

// A fragment, 16x32 bf16 from row-major [m][lda].
DEVI v16bf load_A(const bf16* sA, int row0, int k0, int lda) {
  const int lane = threadIdx.x & 31;
  const bf16* p = sA + (row0 + (lane & 15)) * lda + k0 + 8 * (lane >> 4);
  return frag2x16(p, p + 16);
}

// B fragment, 32x16 bf16 from a TRANSPOSED LDS tile sBt[n][k]; two 16B chunks (k 8h.., 16+8h..).
DEVI v16bf load_Bc(const bf16* sBt, int k0, int n0, int ldb) {
  const int lane = threadIdx.x & 31;
  const bf16* p = sBt + (n0 + (lane & 15)) * ldb + k0 + 8 * (lane >> 4);
  return frag2x16(p, p + 16);
}

DEVI v8f wmma_bf16(v16bf a, v16bf b, v8f c) {
  return __builtin_amdgcn_wmma_f32_16x16x32_bf16(false, a, false, b,
                                                 (short)0, c, false, false);
}

// ---------------------------------------------------------------------------
// Stage 0a: f32 -> bf16 straight conversion (for x)
// ---------------------------------------------------------------------------
__global__ void cvt_bf16_kernel(const float* __restrict__ in,
                                bf16* __restrict__ out, int n4) {
  int i = blockIdx.x * blockDim.x + threadIdx.x;
  if (i < n4) {
    float4 f = ((const float4*)in)[i];
    v4bf o;
    o[0] = (bf16)f.x; o[1] = (bf16)f.y; o[2] = (bf16)f.z; o[3] = (bf16)f.w;
    ((v4bf*)out)[i] = o;
  }
}

// ---------------------------------------------------------------------------
// Stage 0b: f32 -> bf16 conversion WITH transpose (for weights).
// Per z-slice: in [R][C] -> out [C][R].  block = (32,8), tile 32x32 via LDS.
// ---------------------------------------------------------------------------
__global__ __launch_bounds__(256) void cvtT_kernel(const float* __restrict__ in,
                                                   bf16* __restrict__ out,
                                                   int R, int C) {
  __shared__ float t[32][33];
  const float* src = in  + (size_t)blockIdx.z * R * C;
  bf16*        dst = out + (size_t)blockIdx.z * R * C;
  const int r0 = blockIdx.y * 32, c0 = blockIdx.x * 32;
  const int tx = threadIdx.x, ty = threadIdx.y;
#pragma unroll
  for (int i = 0; i < 32; i += 8)
    t[ty + i][tx] = src[(size_t)(r0 + ty + i) * C + c0 + tx];
  __syncthreads();
#pragma unroll
  for (int i = 0; i < 32; i += 8)
    dst[(size_t)(c0 + ty + i) * R + r0 + tx] = (bf16)t[tx][ty + i];
}

// ---------------------------------------------------------------------------
// Stage 1: QKV projection.  grid = (BS/128, NH, 3), block = 128 (4 waves).
// Block tile 128 tokens x 64 head-dims; each wave: 32 rows, 8 WMMA / K-step.
// ---------------------------------------------------------------------------
__global__ __launch_bounds__(128) void qkv_kernel(
    const bf16* __restrict__ x,                                   // [B*S][DM]
    const bf16* __restrict__ wq, const bf16* __restrict__ wk,
    const bf16* __restrict__ wv,                                  // [NH][DH][DM]
    const float* __restrict__ bq, const float* __restrict__ bk,
    const float* __restrict__ bv,                                 // [NH][DH]
    bf16* __restrict__ qo, bf16* __restrict__ ko,                 // [(b*NH+h)*S+s][DH]
    bf16* __restrict__ vo)                                        // [(b*NH+h)*DH+d][S]
{
  constexpr int LDW = 40;                        // sWt [64 n][32 k] + pad
  __shared__ alignas(16) bf16 sX[128 * 32];
  __shared__ alignas(16) bf16 sWt[64 * LDW];

  const int tile = blockIdx.x, h = blockIdx.y, which = blockIdx.z;
  const bf16*  w    = which == 0 ? wq : (which == 1 ? wk : wv);
  const float* bias = which == 0 ? bq : (which == 1 ? bk : bv);

  const int t0 = tile * 128;
  const int lane = threadIdx.x & 31, wave = threadIdx.x >> 5;
  const int wrow0 = wave * 32, hi = lane >> 4, nb = lane & 15;

  v8f acc[2][4] = {};
  for (int k0 = 0; k0 < DM; k0 += 32) {
    __syncthreads();
    copy128x32(x + (size_t)t0 * DM + k0, DM, sX);               // A tile
    copy64x32(w + ((size_t)h * DH) * DM + k0, DM, sWt, LDW);    // B^T tile
    WAIT_ASYNC();
    __syncthreads();
    v16bf a0 = load_A(sX, wrow0, 0, 32);
    v16bf a1 = load_A(sX, wrow0 + 16, 0, 32);
#pragma unroll
    for (int j = 0; j < 4; ++j) {
      v16bf bb = load_Bc(sWt, 0, j * 16, LDW);
      acc[0][j] = wmma_bf16(a0, bb, acc[0][j]);
      acc[1][j] = wmma_bf16(a1, bb, acc[1][j]);
    }
  }

  // epilogue: single base pointer + strides (no per-store branching)
  const int b = t0 / SEQ, s0 = t0 % SEQ;      // 2048 % 128 == 0: one batch/tile
  bf16* outp;
  size_t rs, cs;
  if (which == 2) {  // V transposed: [(b*NH+h)*DH+d][S]
    outp = vo + ((size_t)(b * NH + h) * DH) * SEQ + s0;
    rs = 1; cs = SEQ;
  } else {
    outp = (which == 0 ? qo : ko) + ((size_t)(b * NH + h) * SEQ + s0) * DH;
    rs = DH; cs = 1;
  }
#pragma unroll
  for (int sub = 0; sub < 2; ++sub)
#pragma unroll
    for (int j = 0; j < 4; ++j) {
      const int n = j * 16 + nb;
      const float bia = bias[h * DH + n];
#pragma unroll
      for (int i = 0; i < 8; ++i) {
        const int m = wrow0 + sub * 16 + i + 8 * hi;
        outp[(size_t)m * rs + (size_t)n * cs] = (bf16)(acc[sub][j][i] + bia);
      }
    }
}

// ---------------------------------------------------------------------------
// Stage 2: flash attention.  grid = (S/64, NH, B), block = 128 (4 waves).
// ---------------------------------------------------------------------------
__global__ __launch_bounds__(128) void attn_kernel(
    const bf16* __restrict__ q, const bf16* __restrict__ k,       // [(b*NH+h)*S+s][DH]
    const bf16* __restrict__ vT,                                  // [(b*NH+h)*DH+d][S]
    bf16* __restrict__ z)                                         // [B*S][DM], col h*DH+d
{
  constexpr int LDT = 72;
  __shared__ alignas(16) bf16 sQ[64 * LDT];    // [q row][d]
  __shared__ alignas(16) bf16 sK[64 * LDT];    // [k token][d]  (= B^T for QK^T)
  __shared__ alignas(16) bf16 sVt[64 * LDT];   // [d][k token]  (= B^T for PV)
  __shared__ alignas(16) bf16 sP[64 * LDT];    // [q row][k token] (A for PV)

  const int qt = blockIdx.x, h = blockIdx.y, b = blockIdx.z;
  const size_t head_base = (size_t)(b * NH + h) * SEQ;
  const size_t vt_base   = (size_t)(b * NH + h) * DH * SEQ;

  const int lane = threadIdx.x & 31, wave = threadIdx.x >> 5;
  const int wrow0 = wave * 16, hi = lane >> 4, nb = lane & 15;
  const float scale = 0.125f;  // 1/sqrt(64)

  copy64x64(q + (head_base + (size_t)qt * 64) * DH, DH, sQ);

  float row_max[8], row_sum[8];
#pragma unroll
  for (int i = 0; i < 8; ++i) { row_max[i] = -3.0e38f; row_sum[i] = 0.f; }
  v8f accO[4] = {};

  for (int kt = 0; kt <= qt; ++kt) {
    __syncthreads();
    copy64x64(k + (head_base + (size_t)kt * 64) * DH, DH, sK);
    copy64x64(vT + vt_base + (size_t)kt * 64, SEQ, sVt);
    if (kt < qt) {  // prefetch next K tile into L2 (global_prefetch_b8)
      const bf16* nk = k + (head_base + (size_t)(kt + 1) * 64) * DH;
      __builtin_prefetch(nk + (size_t)threadIdx.x * 32, 0, 1);
    }
    WAIT_ASYNC();
    __syncthreads();

    // S = Q * K^T
    v8f accS[4] = {};
#pragma unroll
    for (int kk = 0; kk < 64; kk += 32) {
      v16bf a = load_A(sQ, wrow0, kk, LDT);
#pragma unroll
      for (int j = 0; j < 4; ++j)
        accS[j] = wmma_bf16(a, load_Bc(sK, kk, j * 16, LDT), accS[j]);
    }

    // scale + causal mask (diagonal tile only)
    const bool diag = (kt == qt);
#pragma unroll
    for (int j = 0; j < 4; ++j) {
      const int col = kt * 64 + j * 16 + nb;
#pragma unroll
      for (int i = 0; i < 8; ++i) {
        float sv = accS[j][i] * scale;
        if (diag) {
          const int row = qt * 64 + wrow0 + i + 8 * hi;
          if (col > row) sv = MASK_NEG;
        }
        accS[j][i] = sv;
      }
    }

    // online softmax (rows live across 16-lane halves of the wave)
#pragma unroll
    for (int i = 0; i < 8; ++i) {
      float mx = accS[0][i];
#pragma unroll
      for (int j = 1; j < 4; ++j) mx = fmaxf(mx, accS[j][i]);
#pragma unroll
      for (int off = 1; off < 16; off <<= 1)
        mx = fmaxf(mx, __shfl_xor(mx, off, 32));
      const float mnew  = fmaxf(row_max[i], mx);
      const float alpha = __expf(row_max[i] - mnew);
      row_max[i] = mnew;
      float psum = 0.f;
#pragma unroll
      for (int j = 0; j < 4; ++j) {
        const float p = __expf(accS[j][i] - mnew);
        accS[j][i] = p;
        psum += p;
      }
#pragma unroll
      for (int off = 1; off < 16; off <<= 1)
        psum += __shfl_xor(psum, off, 32);
      row_sum[i] = row_sum[i] * alpha + psum;
#pragma unroll
      for (int j = 0; j < 4; ++j) accO[j][i] *= alpha;
    }

    // C-layout P -> LDS -> A-layout for the PV matmul
#pragma unroll
    for (int j = 0; j < 4; ++j)
#pragma unroll
      for (int i = 0; i < 8; ++i)
        sP[(wrow0 + i + 8 * hi) * LDT + j * 16 + nb] = (bf16)accS[j][i];
    __syncthreads();

    // O += P * V
#pragma unroll
    for (int kk = 0; kk < 64; kk += 32) {
      v16bf a = load_A(sP, wrow0, kk, LDT);
#pragma unroll
      for (int j = 0; j < 4; ++j)
        accO[j] = wmma_bf16(a, load_Bc(sVt, kk, j * 16, LDT), accO[j]);
    }
  }

  // normalize and emit z (bf16, head-major columns)
#pragma unroll
  for (int j = 0; j < 4; ++j) {
    const int d = j * 16 + nb;
#pragma unroll
    for (int i = 0; i < 8; ++i) {
      const int s = qt * 64 + wrow0 + i + 8 * hi;
      const float o = accO[j][i] / row_sum[i];
      z[((size_t)b * SEQ + s) * DM + h * DH + d] = (bf16)o;
    }
  }
}

// ---------------------------------------------------------------------------
// Stage 3: output projection  out = z @ W_O + b_O  (f32 out).
// grid = (BS/128, DM/64), block = 128.  Block tile 128 x 64.
// ---------------------------------------------------------------------------
__global__ __launch_bounds__(128) void oproj_kernel(
    const bf16* __restrict__ z,    // [B*S][DM]
    const bf16* __restrict__ woT,  // [DM n][DM k]
    const float* __restrict__ bo,  // [DM]
    float* __restrict__ out)       // [B*S][DM]
{
  constexpr int LDW = 40;
  __shared__ alignas(16) bf16 sZ[128 * 32];
  __shared__ alignas(16) bf16 sWt[64 * LDW];

  const int t0 = blockIdx.x * 128, n0 = blockIdx.y * 64;
  const int lane = threadIdx.x & 31, wave = threadIdx.x >> 5;
  const int wrow0 = wave * 32, hi = lane >> 4, nb = lane & 15;

  v8f acc[2][4] = {};
  for (int k0 = 0; k0 < DM; k0 += 32) {
    __syncthreads();
    copy128x32(z + (size_t)t0 * DM + k0, DM, sZ);
    copy64x32(woT + (size_t)n0 * DM + k0, DM, sWt, LDW);
    WAIT_ASYNC();
    __syncthreads();
    v16bf a0 = load_A(sZ, wrow0, 0, 32);
    v16bf a1 = load_A(sZ, wrow0 + 16, 0, 32);
#pragma unroll
    for (int j = 0; j < 4; ++j) {
      v16bf bb = load_Bc(sWt, 0, j * 16, LDW);
      acc[0][j] = wmma_bf16(a0, bb, acc[0][j]);
      acc[1][j] = wmma_bf16(a1, bb, acc[1][j]);
    }
  }

#pragma unroll
  for (int sub = 0; sub < 2; ++sub)
#pragma unroll
    for (int j = 0; j < 4; ++j) {
      const int n = n0 + j * 16 + nb;
      const float bia = bo[n];
#pragma unroll
      for (int i = 0; i < 8; ++i) {
        const int m = t0 + wrow0 + sub * 16 + i + 8 * hi;
        out[(size_t)m * DM + n] = acc[sub][j][i] + bia;
      }
    }
}

// ---------------------------------------------------------------------------
extern "C" void kernel_launch(void* const* d_in, const int* in_sizes, int n_in,
                              void* d_out, int out_size, void* d_ws,
                              size_t ws_size, hipStream_t stream) {
  (void)in_sizes; (void)n_in; (void)out_size; (void)ws_size;
  const float* x  = (const float*)d_in[0];
  const float* Wq = (const float*)d_in[1];
  const float* Wk = (const float*)d_in[2];
  const float* Wv = (const float*)d_in[3];
  const float* Wo = (const float*)d_in[4];
  const float* bq = (const float*)d_in[5];
  const float* bk = (const float*)d_in[6];
  const float* bv = (const float*)d_in[7];
  const float* bo = (const float*)d_in[8];

  char* ws = (char*)d_ws;
  bf16* x16  = (bf16*)(ws + OFF_X);
  bf16* wq16 = (bf16*)(ws + OFF_WQ);
  bf16* wk16 = (bf16*)(ws + OFF_WK);
  bf16* wv16 = (bf16*)(ws + OFF_WV);
  bf16* wo16 = (bf16*)(ws + OFF_WO);
  bf16* q16  = (bf16*)(ws + OFF_Q);
  bf16* k16  = (bf16*)(ws + OFF_K);
  bf16* v16  = (bf16*)(ws + OFF_V);
  bf16* z16  = (bf16*)(ws + OFF_Z);

  const int nX4 = BATCH * SEQ * DM / 4;
  cvt_bf16_kernel<<<(nX4 + 255) / 256, 256, 0, stream>>>(x, x16, nX4);

  dim3 tb(32, 8);
  cvtT_kernel<<<dim3(DH / 32, DM / 32, NH), tb, 0, stream>>>(Wq, wq16, DM, DH);
  cvtT_kernel<<<dim3(DH / 32, DM / 32, NH), tb, 0, stream>>>(Wk, wk16, DM, DH);
  cvtT_kernel<<<dim3(DH / 32, DM / 32, NH), tb, 0, stream>>>(Wv, wv16, DM, DH);
  cvtT_kernel<<<dim3(DM / 32, DM / 32, 1), tb, 0, stream>>>(Wo, wo16, DM, DM);

  qkv_kernel<<<dim3(BATCH * SEQ / 128, NH, 3), 128, 0, stream>>>(
      x16, wq16, wk16, wv16, bq, bk, bv, q16, k16, v16);

  attn_kernel<<<dim3(SEQ / 64, NH, BATCH), 128, 0, stream>>>(q16, k16, v16,
                                                             z16);

  oproj_kernel<<<dim3(BATCH * SEQ / 128, DM / 64), 128, 0, stream>>>(
      z16, wo16, bo, (float*)d_out);
}
